// DecoderLayer_33346126086710
// MI455X (gfx1250) — matrix operand order repaired and checked by reading, not compiled, not run on hardware
//
#include <hip/hip_runtime.h>
#include <cstdint>

// ---------------------------------------------------------------------------
// CDNA5 (gfx1250) types / helpers
// ---------------------------------------------------------------------------
typedef __attribute__((ext_vector_type(16))) __bf16 v16bf;
typedef __attribute__((ext_vector_type(8)))  __bf16 v8bf;
typedef __attribute__((ext_vector_type(4)))  __bf16 v4bf;
typedef __attribute__((ext_vector_type(8)))  float  v8f;

#define WMMA_BF16(A_, B_, C_) \
  __builtin_amdgcn_wmma_f32_16x16x32_bf16(false, (A_), false, (B_), (short)0, (C_), false, false)

// Async global->LDS copy of 16 bytes (ASYNCcnt-tracked, bypasses VGPRs).
__device__ __forceinline__ void async_ld_b128(unsigned lds_off, const void* gaddr) {
  asm volatile("global_load_async_to_lds_b128 %0, %1, off"
               :: "v"(lds_off), "v"(gaddr) : "memory");
}
__device__ __forceinline__ void wait_async0() {
  asm volatile("s_wait_asynccnt 0x0" ::: "memory");
}
__device__ __forceinline__ unsigned lds_off_u32(const void* p) {
  return (unsigned)(uintptr_t)p;  // addrspace(3)->generic keeps LDS byte offset in low 32b
}

// Build a v16bf fragment from two 8-element (16B-aligned) runs (LDS or global).
__device__ __forceinline__ v16bf frag_ld(const __bf16* lo, const __bf16* hi) {
  v8bf a = *(const v8bf*)lo;
  v8bf b = *(const v8bf*)hi;
  return __builtin_shufflevector(a, b, 0,1,2,3,4,5,6,7,8,9,10,11,12,13,14,15);
}

// ---------------------------------------------------------------------------
// Bulk fp32 -> bf16 conversion (vectorized x4).
// ---------------------------------------------------------------------------
__global__ __launch_bounds__(256)
void cvt_bf16_kernel(const float* __restrict__ X, __bf16* __restrict__ Y, int n4) {
  int i = blockIdx.x * 256 + threadIdx.x;
  if (i < n4) {
    float4 t = *(const float4*)(X + (size_t)i * 4);
    v4bf o;
    o[0] = (__bf16)t.x; o[1] = (__bf16)t.y; o[2] = (__bf16)t.z; o[3] = (__bf16)t.w;
    *(v4bf*)(Y + (size_t)i * 4) = o;
  }
}

// ---------------------------------------------------------------------------
// Weight convert + transpose: W[K,N] fp32 -> Wt[N,K] bf16 (32x32 LDS tiles).
// ---------------------------------------------------------------------------
__global__ __launch_bounds__(256)
void wt_t_kernel(const float* __restrict__ W, __bf16* __restrict__ Wt, int K, int N) {
  __shared__ float tile[32][33];
  const int tid = threadIdx.x;
  const int r = tid >> 5, c = tid & 31;
  const int n0 = blockIdx.x * 32, k0 = blockIdx.y * 32;
#pragma unroll
  for (int rr = 0; rr < 4; ++rr)
    tile[c][rr * 8 + r] = W[(size_t)(k0 + rr * 8 + r) * N + (n0 + c)];
  __syncthreads();
#pragma unroll
  for (int rr = 0; rr < 4; ++rr)
    Wt[(size_t)(n0 + rr * 8 + r) * K + (k0 + c)] = (__bf16)tile[rr * 8 + r][c];
}

// ---------------------------------------------------------------------------
// GEMM: acc = A[M,K](bf16) @ Wt[N,K]^T(bf16) + bias. Block tile 128x128, BK=32,
// 8 waves (4 along M x 2 along N), wave tile 32x64 -> 8 WMMAs per K-step.
// Double-buffered LDS fed by async global->LDS copies (software pipeline).
// mode: 0 = f32 out (+optional residual), 1 = bf16 out, 2 = bf16 out * 0.125,
//       3 = bf16 out transposed per head -> [B,H,64,2048] (V operand).
// act:  1 = exact GELU (before residual), applied to f32 accum.
// ---------------------------------------------------------------------------
constexpr int G_LDA = 40, G_LDB = 40;

__device__ __forceinline__ void stage_async(const __bf16* A, const __bf16* Wt,
                                            __bf16* sA, __bf16* sB,
                                            int tid, int m0, int n0, int K, int k0) {
#pragma unroll
  for (int it = 0; it < 2; ++it) {
    int idx = tid + it * 256;          // 0..511
    int row = idx >> 2;                // 0..127
    int c8  = (idx & 3) << 3;          // 0,8,16,24
    async_ld_b128(lds_off_u32(&sA[row * G_LDA + c8]), A  + (size_t)(m0 + row) * K + k0 + c8);
    async_ld_b128(lds_off_u32(&sB[row * G_LDB + c8]), Wt + (size_t)(n0 + row) * K + k0 + c8);
  }
}

__global__ __launch_bounds__(256)
void gemm_bf16_wmma(const __bf16* __restrict__ A, const __bf16* __restrict__ Wt,
                    const float* __restrict__ bias, const float* __restrict__ Res,
                    void* __restrict__ Cp, int M, int N, int K, int mode, int act) {
  constexpr int BM = 128, BN = 128, BK = 32;
  __shared__ __attribute__((aligned(16))) __bf16 As[2][BM * G_LDA];
  __shared__ __attribute__((aligned(16))) __bf16 Bs[2][BN * G_LDB];

  const int tid  = threadIdx.x;
  const int wave = tid >> 5;
  const int lane = tid & 31;
  const int lh   = lane & 15;
  const int hsel = lane >> 4;

  const int m0 = blockIdx.x * BM;
  const int n0 = blockIdx.y * BN;
  const int wm = (wave & 3) * 32;   // 4 waves along M
  const int wn = (wave >> 2) * 64;  // 2 waves along N

  v8f acc[2][4];
#pragma unroll
  for (int mi = 0; mi < 2; ++mi)
#pragma unroll
    for (int ni = 0; ni < 4; ++ni)
#pragma unroll
      for (int j = 0; j < 8; ++j) acc[mi][ni][j] = 0.0f;

  // Prologue: stream tile 0 into buffer 0.
  stage_async(A, Wt, As[0], Bs[0], tid, m0, n0, K, 0);
  wait_async0();
  __syncthreads();

  int cur = 0;
  for (int k0 = 0; k0 < K; k0 += BK) {
    if (k0 + BK < K)  // stream next tile into the other buffer while we compute
      stage_async(A, Wt, As[cur ^ 1], Bs[cur ^ 1], tid, m0, n0, K, k0 + BK);

    // A fragment: row = lane%16, elems 0-7 -> K=8*hsel+i, 8-15 -> K=16+8*hsel+i.
    v16bf af[2];
#pragma unroll
    for (int mi = 0; mi < 2; ++mi) {
      const __bf16* p = &As[cur][(wm + mi * 16 + lh) * G_LDA + hsel * 8];
      af[mi] = frag_ld(p, p + 16);
    }
    // B fragment: col = lane%16, elems 0-7 -> K=8*hsel+i, 8-15 -> K=16+8*hsel+i.
    v16bf bfv[4];
#pragma unroll
    for (int ni = 0; ni < 4; ++ni) {
      const __bf16* p = &Bs[cur][(wn + ni * 16 + lh) * G_LDB + hsel * 8];
      bfv[ni] = frag_ld(p, p + 16);
    }
#pragma unroll
    for (int mi = 0; mi < 2; ++mi)
#pragma unroll
      for (int ni = 0; ni < 4; ++ni)
        acc[mi][ni] = WMMA_BF16(af[mi], bfv[ni], acc[mi][ni]);

    wait_async0();      // next tile fully landed in LDS
    __syncthreads();    // (compiler emits s_wait_dscnt 0 before barrier: reads retired)
    cur ^= 1;
  }

  // Epilogue. C layout: lane holds rows j + 8*hsel, col = lane%16.
#pragma unroll
  for (int mi = 0; mi < 2; ++mi) {
#pragma unroll
    for (int ni = 0; ni < 4; ++ni) {
      const int col = n0 + wn + ni * 16 + lh;
      const float bv = bias[col];
#pragma unroll
      for (int j = 0; j < 8; ++j) {
        const int row = m0 + wm + mi * 16 + j + hsel * 8;
        float v = acc[mi][ni][j] + bv;
        if (act == 1) v = 0.5f * v * (1.0f + erff(v * 0.70710678118654752f));
        if (mode == 0) {
          if (Res) v += Res[(size_t)row * N + col];
          ((float*)Cp)[(size_t)row * N + col] = v;
        } else if (mode == 1) {
          ((__bf16*)Cp)[(size_t)row * N + col] = (__bf16)v;
        } else if (mode == 2) {
          ((__bf16*)Cp)[(size_t)row * N + col] = (__bf16)(v * 0.125f);
        } else {  // mode 3: V operand, per-head transpose [B,H,64,Lseq], Lseq=2048
          const int bb = row >> 11, t = row & 2047;
          const int hh = col >> 6,  dk = col & 63;
          ((__bf16*)Cp)[(((size_t)bb * 16 + hh) * 64 + dk) * 2048 + t] = (__bf16)v;
        }
      }
    }
  }
}

// ---------------------------------------------------------------------------
// Flash attention. Q[B,L,D] bf16 (pre-scaled by 1/8), K[B,L,D] bf16,
// Vt[B,H,64,Lk] bf16 (pre-transposed). One wave per 32-row Q tile (two 16-row
// WMMA row-tiles reuse every K/V fragment), 32-key chunks -> 16 WMMAs/chunk.
// Only the 32x32 P tile round-trips per-wave LDS (C-layout -> A-layout).
// ---------------------------------------------------------------------------
constexpr int A_LDP = 40;

__device__ __forceinline__ void online_softmax(
    v8f& s0, v8f& s1, v8f o[4], float m_[8], float l_[8],
    __bf16* Prow, int qbase, int k0, int lh, int hsel, int causal) {
#pragma unroll
  for (int j = 0; j < 8; ++j) {
    float a0 = s0[j], a1 = s1[j];
    if (causal) {
      const int qrow = qbase + j + hsel * 8;
      if (k0 + lh      > qrow) a0 = -1e30f;
      if (k0 + 16 + lh > qrow) a1 = -1e30f;
    }
    float mt = fmaxf(a0, a1);
    mt = fmaxf(mt, __shfl_xor(mt, 1, 32));
    mt = fmaxf(mt, __shfl_xor(mt, 2, 32));
    mt = fmaxf(mt, __shfl_xor(mt, 4, 32));
    mt = fmaxf(mt, __shfl_xor(mt, 8, 32));
    const float mn    = fmaxf(m_[j], mt);
    const float alpha = __expf(m_[j] - mn);
    const float p0 = __expf(a0 - mn);
    const float p1 = __expf(a1 - mn);
    float rs = p0 + p1;
    rs += __shfl_xor(rs, 1, 32);
    rs += __shfl_xor(rs, 2, 32);
    rs += __shfl_xor(rs, 4, 32);
    rs += __shfl_xor(rs, 8, 32);
    l_[j] = l_[j] * alpha + rs;
    m_[j] = mn;
    o[0][j] *= alpha; o[1][j] *= alpha; o[2][j] *= alpha; o[3][j] *= alpha;
    const int prow = j + hsel * 8;
    Prow[prow * A_LDP + lh]      = (__bf16)p0;
    Prow[prow * A_LDP + 16 + lh] = (__bf16)p1;
  }
}

__global__ __launch_bounds__(256)
void attn_fa_wmma(const __bf16* __restrict__ Q, const __bf16* __restrict__ K,
                  const __bf16* __restrict__ Vt, __bf16* __restrict__ O,
                  int Lq, int Lk, int causal) {
  constexpr int Dm = 1024;
  __shared__ __attribute__((aligned(16))) __bf16 smP[8 * 32 * A_LDP];

  const int tid = threadIdx.x, wave = tid >> 5, lane = tid & 31;
  const int lh = lane & 15, hsel = lane >> 4;
  const int h = blockIdx.y, b = blockIdx.z;
  const int q0 = (blockIdx.x * 8 + wave) * 32;
  __bf16* Pb = &smP[wave * 32 * A_LDP];   // two 16x32 row-tiles

  const __bf16* Qb  = Q  + ((size_t)b * Lq) * Dm + h * 64;
  const __bf16* Kb  = K  + ((size_t)b * Lk) * Dm + h * 64;
  const __bf16* Vtb = Vt + ((size_t)b * 16 + h) * (size_t)64 * Lk;

  // Q fragments: [row-tile r][K-frag f], pre-scaled by 1/sqrt(64).
  v16bf qa[2][2];
#pragma unroll
  for (int r = 0; r < 2; ++r)
#pragma unroll
    for (int f = 0; f < 2; ++f) {
      const __bf16* p = Qb + (size_t)(q0 + r * 16 + lh) * Dm + f * 32 + hsel * 8;
      qa[r][f] = frag_ld(p, p + 16);
    }

  v8f o[2][4];
  float m_[2][8], l_[2][8];
#pragma unroll
  for (int r = 0; r < 2; ++r) {
#pragma unroll
    for (int n = 0; n < 4; ++n)
#pragma unroll
      for (int j = 0; j < 8; ++j) o[r][n][j] = 0.0f;
#pragma unroll
    for (int j = 0; j < 8; ++j) { m_[r][j] = -1e30f; l_[r][j] = 0.0f; }
  }

  const int kend = causal ? (q0 + 32) : Lk;
  for (int k0 = 0; k0 < kend; k0 += 32) {
    // S[2][16x32] = Q @ K_chunk^T; each K fragment feeds both row-tiles.
    v8f s[2][2];
#pragma unroll
    for (int r = 0; r < 2; ++r)
#pragma unroll
      for (int j = 0; j < 8; ++j) { s[r][0][j] = 0.0f; s[r][1][j] = 0.0f; }
#pragma unroll
    for (int f = 0; f < 2; ++f) {
      const __bf16* pk0 = Kb + (size_t)(k0 + lh) * Dm + f * 32 + hsel * 8;
      const __bf16* pk1 = pk0 + (size_t)16 * Dm;
      v16bf kb0 = frag_ld(pk0, pk0 + 16);
      v16bf kb1 = frag_ld(pk1, pk1 + 16);
#pragma unroll
      for (int r = 0; r < 2; ++r) {
        s[r][0] = WMMA_BF16(qa[r][f], kb0, s[r][0]);
        s[r][1] = WMMA_BF16(qa[r][f], kb1, s[r][1]);
      }
    }

#pragma unroll
    for (int r = 0; r < 2; ++r)
      online_softmax(s[r][0], s[r][1], o[r], m_[r], l_[r],
                     Pb + r * 16 * A_LDP, q0 + r * 16, k0, lh, hsel, causal);

    // O[2][16x64] += P @ V; V fragments shared by both row-tiles.
    v16bf pa[2];
#pragma unroll
    for (int r = 0; r < 2; ++r) {
      const __bf16* pp = &Pb[(r * 16 + lh) * A_LDP + hsel * 8];
      pa[r] = frag_ld(pp, pp + 16);   // same-wave LDS is in-order: no barrier
    }
#pragma unroll
    for (int n = 0; n < 4; ++n) {
      const __bf16* pv = Vtb + (size_t)(n * 16 + lh) * Lk + k0 + hsel * 8;
      v16bf vb = frag_ld(pv, pv + 16);
      o[0][n] = WMMA_BF16(pa[0], vb, o[0][n]);
      o[1][n] = WMMA_BF16(pa[1], vb, o[1][n]);
    }
  }

  // Normalize, write bf16 concat-head output [B, Lq, D].
#pragma unroll
  for (int r = 0; r < 2; ++r)
#pragma unroll
    for (int j = 0; j < 8; ++j) {
      const float inv = 1.0f / l_[r][j];
      const int row = q0 + r * 16 + j + hsel * 8;
      __bf16* po = O + ((size_t)b * Lq + row) * Dm + h * 64 + lh;
      po[0]  = (__bf16)(o[r][0][j] * inv);
      po[16] = (__bf16)(o[r][1][j] * inv);
      po[32] = (__bf16)(o[r][2][j] * inv);
      po[48] = (__bf16)(o[r][3][j] * inv);
    }
}

// ---------------------------------------------------------------------------
// LayerNorm for D=1024: one block per row, one float4 per thread, eps 1e-5.
// Writes f32 (and optional bf16 copy).
// ---------------------------------------------------------------------------
__global__ __launch_bounds__(256)
void ln_kernel(const float* __restrict__ X, const float* __restrict__ g,
               const float* __restrict__ be, float* __restrict__ Y,
               __bf16* __restrict__ Yb) {
  constexpr int Dn = 1024;
  __shared__ float rs[8], rq[8];
  const int tid = threadIdx.x, wave = tid >> 5, lane = tid & 31;
  const size_t base = (size_t)blockIdx.x * Dn;
  float4 t = *(const float4*)(X + base + tid * 4);
  float s = t.x + t.y + t.z + t.w;
  float q = t.x * t.x + t.y * t.y + t.z * t.z + t.w * t.w;
#pragma unroll
  for (int off = 1; off < 32; off <<= 1) {
    s += __shfl_xor(s, off, 32);
    q += __shfl_xor(q, off, 32);
  }
  if (lane == 0) { rs[wave] = s; rq[wave] = q; }
  __syncthreads();
  float ts = 0.0f, tq = 0.0f;
#pragma unroll
  for (int i = 0; i < 8; ++i) { ts += rs[i]; tq += rq[i]; }
  const float mu  = ts * (1.0f / Dn);
  const float var = tq * (1.0f / Dn) - mu * mu;
  const float rst = rsqrtf(var + 1e-5f);
  float4 gv = *(const float4*)(g  + tid * 4);
  float4 bv = *(const float4*)(be + tid * 4);
  float4 ov;
  ov.x = (t.x - mu) * rst * gv.x + bv.x;
  ov.y = (t.y - mu) * rst * gv.y + bv.y;
  ov.z = (t.z - mu) * rst * gv.z + bv.z;
  ov.w = (t.w - mu) * rst * gv.w + bv.w;
  *(float4*)(Y + base + tid * 4) = ov;
  if (Yb) {
    v4bf o;
    o[0] = (__bf16)ov.x; o[1] = (__bf16)ov.y; o[2] = (__bf16)ov.z; o[3] = (__bf16)ov.w;
    *(v4bf*)(Yb + base + tid * 4) = o;
  }
}

// ---------------------------------------------------------------------------
// Orchestration
// ---------------------------------------------------------------------------
extern "C" void kernel_launch(void* const* d_in, const int* in_sizes, int n_in,
                              void* d_out, int out_size, void* d_ws, size_t ws_size,
                              hipStream_t stream) {
  (void)in_sizes; (void)n_in; (void)out_size; (void)ws_size;

  const float* x   = (const float*)d_in[0];
  const float* enc = (const float*)d_in[1];
  // d_in[2] tgt_mask (tril) and d_in[3] src_mask (all-ones) handled analytically.
  const float* wsrc[10] = {
    (const float*)d_in[4],  (const float*)d_in[6],  (const float*)d_in[8],
    (const float*)d_in[10], (const float*)d_in[12], (const float*)d_in[14],
    (const float*)d_in[16], (const float*)d_in[18], (const float*)d_in[20],
    (const float*)d_in[22]
  };
  const float* sbq = (const float*)d_in[5];   const float* sbk = (const float*)d_in[7];
  const float* sbv = (const float*)d_in[9];   const float* sbo = (const float*)d_in[11];
  const float* cbq = (const float*)d_in[13];  const float* cbk = (const float*)d_in[15];
  const float* cbv = (const float*)d_in[17];  const float* cbo = (const float*)d_in[19];
  const float* fb1 = (const float*)d_in[21];  const float* fb2 = (const float*)d_in[23];
  const float* g1  = (const float*)d_in[24];  const float* b1  = (const float*)d_in[25];
  const float* g2  = (const float*)d_in[26];  const float* b2  = (const float*)d_in[27];
  const float* g3  = (const float*)d_in[28];  const float* b3  = (const float*)d_in[29];

  constexpr int Bb = 2, LT = 2048, LS = 2048, Dm = 1024, DFF = 4096;
  const int M = Bb * LT;                        // 4096 tokens
  const size_t S1 = (size_t)M * Dm;             // 4M elements

  // Workspace carve-up (256B aligned).
  char* wp = (char*)d_ws;
  auto carve = [&wp](size_t bytes) { char* r = wp; wp += (bytes + 255) & ~(size_t)255; return r; };
  __bf16* wt[10];
  for (int i = 0; i < 10; ++i) {
    size_t elems = (i < 8) ? (size_t)Dm * Dm : (size_t)Dm * DFF;
    wt[i] = (__bf16*)carve(elems * sizeof(__bf16));
  }
  __bf16* x_bf   = (__bf16*)carve(S1 * 2);
  __bf16* enc_bf = (__bf16*)carve(S1 * 2);
  __bf16* qb     = (__bf16*)carve(S1 * 2);
  __bf16* kb     = (__bf16*)carve(S1 * 2);
  __bf16* vtb    = (__bf16*)carve(S1 * 2);      // [B,H,64,Lseq]
  __bf16* attnb  = (__bf16*)carve(S1 * 2);
  __bf16* xn1b   = (__bf16*)carve(S1 * 2);
  __bf16* xn2b   = (__bf16*)carve(S1 * 2);
  __bf16* hb     = (__bf16*)carve((size_t)M * DFF * 2);
  float*  x1     = (float*)carve(S1 * 4);
  float*  xn1    = (float*)carve(S1 * 4);
  float*  xn2    = (float*)carve(S1 * 4);

  dim3 blk(256, 1, 1);
  dim3 gP (M / 128, Dm  / 128, 1);
  dim3 gF1(M / 128, DFF / 128, 1);
  dim3 gA (LT / 256, 16, Bb);
  dim3 gLN(M, 1, 1);
  dim3 gCV((int)(S1 / 4 / 256), 1, 1);

  // ---- one-time conversions ----
  cvt_bf16_kernel<<<gCV, blk, 0, stream>>>(x,   x_bf,   (int)(S1 / 4));
  cvt_bf16_kernel<<<gCV, blk, 0, stream>>>(enc, enc_bf, (int)(S1 / 4));
  for (int i = 0; i < 8; ++i)
    wt_t_kernel<<<dim3(Dm / 32, Dm / 32), blk, 0, stream>>>(wsrc[i], wt[i], Dm, Dm);
  wt_t_kernel<<<dim3(DFF / 32, Dm / 32), blk, 0, stream>>>(wsrc[8], wt[8], Dm, DFF);  // fw1
  wt_t_kernel<<<dim3(Dm / 32, DFF / 32), blk, 0, stream>>>(wsrc[9], wt[9], DFF, Dm);  // fw2

  // ---- self attention (causal) ----
  gemm_bf16_wmma<<<gP, blk, 0, stream>>>(x_bf, wt[0], sbq, nullptr, qb,  M, Dm, Dm, 2, 0);
  gemm_bf16_wmma<<<gP, blk, 0, stream>>>(x_bf, wt[1], sbk, nullptr, kb,  M, Dm, Dm, 1, 0);
  gemm_bf16_wmma<<<gP, blk, 0, stream>>>(x_bf, wt[2], sbv, nullptr, vtb, M, Dm, Dm, 3, 0);
  attn_fa_wmma<<<gA, blk, 0, stream>>>(qb, kb, vtb, attnb, LT, LT, 1);
  gemm_bf16_wmma<<<gP, blk, 0, stream>>>(attnb, wt[3], sbo, x, x1, M, Dm, Dm, 0, 0);
  ln_kernel<<<gLN, blk, 0, stream>>>(x1, g1, b1, xn1, xn1b);

  // ---- cross attention (no mask) ----
  gemm_bf16_wmma<<<gP, blk, 0, stream>>>(xn1b,   wt[4], cbq, nullptr, qb,  M, Dm, Dm, 2, 0);
  gemm_bf16_wmma<<<gP, blk, 0, stream>>>(enc_bf, wt[5], cbk, nullptr, kb,  M, Dm, Dm, 1, 0);
  gemm_bf16_wmma<<<gP, blk, 0, stream>>>(enc_bf, wt[6], cbv, nullptr, vtb, M, Dm, Dm, 3, 0);
  attn_fa_wmma<<<gA, blk, 0, stream>>>(qb, kb, vtb, attnb, LT, LS, 0);
  gemm_bf16_wmma<<<gP, blk, 0, stream>>>(attnb, wt[7], cbo, xn1, x1, M, Dm, Dm, 0, 0);
  ln_kernel<<<gLN, blk, 0, stream>>>(x1, g2, b2, xn2, xn2b);

  // ---- FFN ----
  gemm_bf16_wmma<<<gF1, blk, 0, stream>>>(xn2b, wt[8], fb1, nullptr, hb, M, DFF, Dm, 1, 1);
  gemm_bf16_wmma<<<gP,  blk, 0, stream>>>(hb,   wt[9], fb2, xn2,     x1, M, Dm, DFF, 0, 0);
  ln_kernel<<<gLN, blk, 0, stream>>>(x1, g3, b3, (float*)d_out, nullptr);
}
